// CharBiLSTMCRF_16853451670187
// MI455X (gfx1250) — matrix operand order repaired and checked by reading, not compiled, not run on hardware
//
#include <hip/hip_runtime.h>
#include <hip/hip_bf16.h>
#include <math.h>

typedef __attribute__((ext_vector_type(16))) _Float16 v16h;
typedef __attribute__((ext_vector_type(8)))  _Float16 v8h;
typedef __attribute__((ext_vector_type(8)))  float    v8f;
typedef __attribute__((ext_vector_type(4)))  unsigned int v4u;
typedef __attribute__((ext_vector_type(8)))  int      v8i;
typedef __attribute__((ext_vector_type(4)))  int      v4i;

#define S_LEN 1024
#define B_N   128
#define E_DIM 128
#define H_DIM 128
#define G4H   512     // 4*H
#define K_TAGS 15
#define K_PAD  16

#if __has_builtin(__builtin_amdgcn_tensor_load_to_lds) && __has_builtin(__builtin_amdgcn_s_wait_tensorcnt)
#define USE_TDM 1
#else
#define USE_TDM 0
#endif

// ---------------- WMMA operand loaders (16x16x32 f16) ----------------
// A-matrix 16x32: lane m = lane%16, hs = lane/16.
//   a[0..7]  = row[hs*8 + 0..7], a[8..15] = row[16 + hs*8 + 0..7]
__device__ __forceinline__ v16h load_a_tile(const _Float16* row, int hs) {
    v8h lo = *(const v8h*)(row + hs * 8);
    v8h hi = *(const v8h*)(row + 16 + hs * 8);
    v16h r;
#pragma unroll
    for (int i = 0; i < 8; ++i) { r[i] = lo[i]; r[i + 8] = hi[i]; }
    return r;
}
// B-matrix 32x16: lane n = lane%16, hs = lane/16. b[0..7] = Wrow[hs*8 + 0..7], b[8..15] = Wrow[16 + hs*8 + 0..7]
__device__ __forceinline__ v16h load_b_tile(const _Float16* wrow, int hs) {
    v8h lo = *(const v8h*)(wrow + hs * 8);
    v8h hi = *(const v8h*)(wrow + 16 + hs * 8);
    v16h r;
#pragma unroll
    for (int i = 0; i < 8; ++i) { r[i] = lo[i]; r[i + 8] = hi[i]; }
    return r;
}

__device__ __forceinline__ float sigmoidf_(float x) { return 1.0f / (1.0f + expf(-x)); }

#if USE_TDM
// Issue a TDM load of a [rows=128, cols=in_dim] f16 tile (contiguous, row stride
// = in_dim elements) from global memory into LDS at byte offset lds_off.
// D# layout per CDNA5 ISA (2-D tensor, groups 2/3 zero).
// Toolchain uses the 6-arg builtin form: (g0 u32x4, g1 i32x8, i32x4, i32x4, i32x8, cpol).
__device__ __forceinline__ void tdm_load_x(const _Float16* gsrc, unsigned lds_off, int in_dim) {
    unsigned long long ga = (unsigned long long)(size_t)gsrc;
    v4u g0;
    g0[0] = 1u;                                               // count=1, user mode
    g0[1] = lds_off;                                          // lds_addr (bytes)
    g0[2] = (unsigned)(ga & 0xFFFFFFFFu);                     // global_addr[31:0]
    g0[3] = (unsigned)((ga >> 32) & 0x01FFFFFFu) | (2u << 30);// global_addr[56:32] | type=2
    v8i g1;
    g1[0] = 0x00010000;                   // workgroup_mask=0, data_size=1 (2 bytes)
    g1[1] = (in_dim & 0xFFFF) << 16;      // tensor_dim0[15:0] in bits[63:48]
    g1[2] = (B_N << 16);                  // tensor_dim0[31:16]=0 | tensor_dim1[15:0]=128
    g1[3] = (in_dim << 16);               // tensor_dim1[31:16]=0 | tile_dim0=in_dim
    g1[4] = B_N;                          // tile_dim1=128, tile_dim2=0
    g1[5] = in_dim;                       // tensor_dim0_stride[31:0]
    g1[6] = 0;                            // stride hi, tensor_dim1_stride lo
    g1[7] = 0;
    v4i z4 = {0, 0, 0, 0};
    v8i z8 = {0, 0, 0, 0, 0, 0, 0, 0};
    __builtin_amdgcn_tensor_load_to_lds(g0, g1, z4, z4, z8, 0);
}
#endif

// ---------------- prep kernels ----------------
__global__ void cvt_f16_k(const float* __restrict__ s, _Float16* __restrict__ d, int n) {
    int i = blockIdx.x * blockDim.x + threadIdx.x;
    if (i < n) d[i] = (_Float16)s[i];
}

__global__ void bias_sum_k(const float* __restrict__ a, const float* __restrict__ b,
                           float* __restrict__ d, int n) {
    int i = blockIdx.x * blockDim.x + threadIdx.x;
    if (i < n) d[i] = a[i] + b[i];
}

__global__ void fcpad_k(const float* __restrict__ w, const float* __restrict__ bv,
                        _Float16* __restrict__ w16, float* __restrict__ b16) {
    int i = blockIdx.x * blockDim.x + threadIdx.x;
    if (i < K_PAD * 2 * H_DIM) {
        int r = i >> 8, c = i & 255;
        w16[i] = (r < K_TAGS) ? (_Float16)w[r * 2 * H_DIM + c] : (_Float16)0.0f;
    }
    if (i < K_PAD) b16[i] = (i < K_TAGS) ? bv[i] : 0.0f;
}

// char_ids [B,S] -> x f16 [S,B,E]
__global__ void embed_k(const int* __restrict__ ids, const float* __restrict__ emb,
                        _Float16* __restrict__ x16) {
    int idx = blockIdx.x * blockDim.x + threadIdx.x;   // (s*B + b)*E + e
    int e = idx & (E_DIM - 1);
    int b = (idx >> 7) & (B_N - 1);
    int s = idx >> 14;
    int id = ids[b * S_LEN + s];
    x16[idx] = (_Float16)emb[id * E_DIM + e];
}

// ---------------- recurrent BiLSTM layer ----------------
// grid = 2 (dir), block = 1024 (32 waves). Wave w owns gate columns [16w, 16w+16).
// LDS: Gh [128][512] f16 (128 KB) + hsh [128][128] f16 (32 KB) + 2 x-tile buffers.
template <int IN>
__global__ __launch_bounds__(1024) void lstm_k(
    const _Float16* __restrict__ x,                      // [S, B, IN]
    const _Float16* __restrict__ wih0, const _Float16* __restrict__ whh0,
    const float*    __restrict__ bias0,
    const _Float16* __restrict__ wih1, const _Float16* __restrict__ whh1,
    const float*    __restrict__ bias1,
    _Float16* __restrict__ hout)                         // [S, B, 2H]
{
    extern __shared__ char smem[];
    _Float16* Gh  = (_Float16*)smem;                         // 128 x 512 f16
    _Float16* hsh = (_Float16*)(smem + B_N * G4H * 2);       // 128 x 128 f16
    const unsigned xoff0 = B_N * G4H * 2 + B_N * H_DIM * 2;  // x double buffer
    const unsigned xoff1 = xoff0 + B_N * IN * 2;
    _Float16* xb0 = (_Float16*)(smem + xoff0);
    _Float16* xb1 = (_Float16*)(smem + xoff1);

    const int dir = blockIdx.x;                          // 0 = fwd, 1 = bwd
    const _Float16* wih  = dir ? wih1  : wih0;
    const _Float16* whh  = dir ? whh1  : whh0;
    const float*    bias = dir ? bias1 : bias0;

    const int tid  = threadIdx.x;
    const int wave = tid >> 5;
    const int lane = tid & 31;
    const int hs   = lane >> 4;
    const int ln16 = lane & 15;
    const int n0   = wave * 16;                          // gate-column tile base

    // B operands (weights) for this wave's n-tile: invariant across all steps.
    constexpr int KC_IH = IN / 32;
    v16h bw_ih[KC_IH];
#pragma unroll
    for (int kc = 0; kc < KC_IH; ++kc)
        bw_ih[kc] = load_b_tile(wih + (size_t)(n0 + ln16) * IN + kc * 32, hs);
    v16h bw_hh[4];
#pragma unroll
    for (int kc = 0; kc < 4; ++kc)
        bw_hh[kc] = load_b_tile(whh + (size_t)(n0 + ln16) * H_DIM + kc * 32, hs);
    const float bias_n = bias[n0 + ln16];

    // c-state: thread owns (b = tid/8, h = (tid%8)*16 + j), j=0..15
    const int cb  = tid >> 3;
    const int ch0 = (tid & 7) << 4;
    float creg[16];
#pragma unroll
    for (int j = 0; j < 16; ++j) creg[j] = 0.0f;
#pragma unroll
    for (int j = 0; j < 16; ++j) hsh[cb * H_DIM + ch0 + j] = (_Float16)0.0f;

#if USE_TDM
    // Prologue: DMA the first time-step's x tile into buffer 0.
    if (wave == 0) {
        const int s0 = dir ? (S_LEN - 1) : 0;
        tdm_load_x(x + (size_t)s0 * B_N * IN, xoff0, IN);
    }
    __builtin_amdgcn_s_wait_tensorcnt((short)0);
#endif
    __syncthreads();

    for (int t = 0; t < S_LEN; ++t) {
        const int s = dir ? (S_LEN - 1 - t) : t;
#if USE_TDM
        // Prefetch next step's x tile into the other buffer (overlaps compute).
        if (wave == 0 && (t + 1) < S_LEN) {
            const int sn = dir ? (S_LEN - 2 - t) : (t + 1);
            tdm_load_x(x + (size_t)sn * B_N * IN, ((t + 1) & 1) ? xoff1 : xoff0, IN);
        }
        const _Float16* xs = (t & 1) ? xb1 : xb0;
#else
        const _Float16* xs = x + (size_t)s * B_N * IN;
#endif

        for (int bt = 0; bt < 8; ++bt) {                 // 8 batch tiles of 16
            v8f acc;
#pragma unroll
            for (int v = 0; v < 8; ++v) acc[v] = bias_n;
            const _Float16* arow = xs + (size_t)(bt * 16 + ln16) * IN;
#pragma unroll
            for (int kc = 0; kc < KC_IH; ++kc) {
                v16h a = load_a_tile(arow + kc * 32, hs);
                acc = __builtin_amdgcn_wmma_f32_16x16x32_f16(
                    false, a, false, bw_ih[kc], (short)0, acc, false, false);
            }
            const _Float16* hrow = hsh + (bt * 16 + ln16) * H_DIM;
#pragma unroll
            for (int kc = 0; kc < 4; ++kc) {
                v16h a = load_a_tile(hrow + kc * 32, hs);
                acc = __builtin_amdgcn_wmma_f32_16x16x32_f16(
                    false, a, false, bw_hh[kc], (short)0, acc, false, false);
            }
#pragma unroll
            for (int v = 0; v < 8; ++v)
                Gh[(bt * 16 + v + 8 * hs) * G4H + n0 + ln16] = (_Float16)acc[v];
        }
        __syncthreads();

        // gate nonlinearity + state update (i,f,g,o split along 512)
        const _Float16* gb = Gh + cb * G4H;
#pragma unroll
        for (int j = 0; j < 16; ++j) {
            int h = ch0 + j;
            float ig = sigmoidf_((float)gb[h]);
            float fg = sigmoidf_((float)gb[128 + h]);
            float gg = tanhf((float)gb[256 + h]);
            float og = sigmoidf_((float)gb[384 + h]);
            float c  = fg * creg[j] + ig * gg;
            creg[j]  = c;
            _Float16 hh = (_Float16)(og * tanhf(c));
            hsh[cb * H_DIM + h] = hh;
            hout[((size_t)s * B_N + cb) * (2 * H_DIM) + dir * H_DIM + h] = hh;
        }
#if USE_TDM
        __builtin_amdgcn_s_wait_tensorcnt((short)0);  // next x tile resident
#endif
        __syncthreads();
    }
}

// ---------------- emissions GEMM: [S*B, 256] x [256, 16] ----------------
__global__ __launch_bounds__(256) void emis_k(const _Float16* __restrict__ h1,
                                              const _Float16* __restrict__ fcw,  // [16,256]
                                              const float* __restrict__ fcb,     // [16]
                                              float* __restrict__ em) {          // [S*B,16]
    int gwave = (blockIdx.x * blockDim.x + threadIdx.x) >> 5;
    int lane  = threadIdx.x & 31;
    int hs = lane >> 4, ln16 = lane & 15;
    size_t m0 = (size_t)gwave * 16;
    v8f acc;
    float bn = fcb[ln16];
#pragma unroll
    for (int v = 0; v < 8; ++v) acc[v] = bn;
    const _Float16* arow = h1 + (m0 + ln16) * (2 * H_DIM);
    const _Float16* brow = fcw + ln16 * (2 * H_DIM);
#pragma unroll
    for (int kc = 0; kc < 8; ++kc) {
        v16h a = load_a_tile(arow + kc * 32, hs);
        v16h b = load_b_tile(brow + kc * 32, hs);
        acc = __builtin_amdgcn_wmma_f32_16x16x32_f16(
            false, a, false, b, (short)0, acc, false, false);
    }
#pragma unroll
    for (int v = 0; v < 8; ++v)
        em[(m0 + v + 8 * hs) * K_PAD + ln16] = acc[v];
}

// ---------------- CRF NLL: one block, thread = batch element ----------------
__global__ __launch_bounds__(128) void crf_k(const float* __restrict__ em,       // [S,B,16]
                                             const int* __restrict__ tags,        // [B,S]
                                             const unsigned char* __restrict__ mask,
                                             const float* __restrict__ trans,     // [15,15]
                                             const float* __restrict__ startv,
                                             const float* __restrict__ endv,
                                             float* __restrict__ out) {
    __shared__ float tr[K_TAGS * K_TAGS], st[K_TAGS], en[K_TAGS], red[128];
    int tid = threadIdx.x;
    for (int i = tid; i < K_TAGS * K_TAGS; i += 128) tr[i] = trans[i];
    if (tid < K_TAGS) { st[tid] = startv[tid]; en[tid] = endv[tid]; }
    __syncthreads();

    int b = tid;
    int prev = tags[b * S_LEN + 0];
    float num = st[prev] + em[(size_t)b * K_PAD + prev];

    float alpha[K_TAGS];
#pragma unroll
    for (int k = 0; k < K_TAGS; ++k) alpha[k] = st[k] + em[(size_t)b * K_PAD + k];

    for (int s = 1; s < S_LEN; ++s) {
        const float* ems = em + ((size_t)s * B_N + b) * K_PAD;
        float m = mask[b * S_LEN + s] ? 1.0f : 0.0f;
        int tg = tags[b * S_LEN + s];
        num += (tr[prev * K_TAGS + tg] + ems[tg]) * m;
        if (m > 0.0f) prev = tg;

        float na[K_TAGS];
        for (int j = 0; j < K_TAGS; ++j) {
            float mx = -3.4e38f;
            for (int i = 0; i < K_TAGS; ++i) mx = fmaxf(mx, alpha[i] + tr[i * K_TAGS + j]);
            float ss = 0.0f;
            for (int i = 0; i < K_TAGS; ++i) ss += expf(alpha[i] + tr[i * K_TAGS + j] - mx);
            na[j] = mx + logf(ss) + ems[j];
        }
        if (m > 0.0f) {
            for (int j = 0; j < K_TAGS; ++j) alpha[j] = na[j];
        }
    }
    num += en[prev];
    float mx = -3.4e38f;
    for (int k = 0; k < K_TAGS; ++k) mx = fmaxf(mx, alpha[k] + en[k]);
    float ss = 0.0f;
    for (int k = 0; k < K_TAGS; ++k) ss += expf(alpha[k] + en[k] - mx);
    float logZ = mx + logf(ss);

    red[tid] = logZ - num;
    __syncthreads();
    for (int off = 64; off > 0; off >>= 1) {
        if (tid < off) red[tid] += red[tid + off];
        __syncthreads();
    }
    if (tid == 0) out[0] = red[0];
}

// ---------------- host launch ----------------
extern "C" void kernel_launch(void* const* d_in, const int* in_sizes, int n_in,
                              void* d_out, int out_size, void* d_ws, size_t ws_size,
                              hipStream_t stream) {
    (void)in_sizes; (void)n_in; (void)out_size; (void)ws_size;

    const int*           char_ids = (const int*)d_in[0];
    const int*           tags     = (const int*)d_in[1];
    const unsigned char* mask     = (const unsigned char*)d_in[2];
    const float*         emb      = (const float*)d_in[3];
    const float* wih[2][2] = { {(const float*)d_in[4],  (const float*)d_in[8]},
                               {(const float*)d_in[12], (const float*)d_in[16]} };
    const float* whh[2][2] = { {(const float*)d_in[5],  (const float*)d_in[9]},
                               {(const float*)d_in[13], (const float*)d_in[17]} };
    const float* bih[2][2] = { {(const float*)d_in[6],  (const float*)d_in[10]},
                               {(const float*)d_in[14], (const float*)d_in[18]} };
    const float* bhh[2][2] = { {(const float*)d_in[7],  (const float*)d_in[11]},
                               {(const float*)d_in[15], (const float*)d_in[19]} };
    const float* fcw    = (const float*)d_in[20];
    const float* fcb    = (const float*)d_in[21];
    const float* transm = (const float*)d_in[22];
    const float* startv = (const float*)d_in[23];
    const float* endv   = (const float*)d_in[24];

    char* ws = (char*)d_ws;
    size_t off = 0;
    auto alloc = [&](size_t bytes) -> char* {
        char* p = ws + off;
        off += (bytes + 255) & ~(size_t)255;
        return p;
    };

    _Float16* X16 = (_Float16*)alloc((size_t)S_LEN * B_N * E_DIM * 2);
    _Float16* H0  = (_Float16*)alloc((size_t)S_LEN * B_N * 2 * H_DIM * 2);
    _Float16* H1  = (_Float16*)alloc((size_t)S_LEN * B_N * 2 * H_DIM * 2);
    float*    EM  = (float*)  alloc((size_t)S_LEN * B_N * K_PAD * 4);

    _Float16* WIH16[2][2]; _Float16* WHH16[2][2]; float* BIAS[2][2];
    for (int l = 0; l < 2; ++l)
        for (int d = 0; d < 2; ++d) {
            int in_dim = l ? 2 * H_DIM : E_DIM;
            WIH16[l][d] = (_Float16*)alloc((size_t)G4H * in_dim * 2);
            WHH16[l][d] = (_Float16*)alloc((size_t)G4H * H_DIM * 2);
            BIAS[l][d]  = (float*)alloc(G4H * 4);
        }
    _Float16* FCW16 = (_Float16*)alloc(K_PAD * 2 * H_DIM * 2);
    float*    FCB16 = (float*)alloc(K_PAD * 4);

    // weight conversion f32 -> f16, bias folding
    for (int l = 0; l < 2; ++l)
        for (int d = 0; d < 2; ++d) {
            int in_dim = l ? 2 * H_DIM : E_DIM;
            int n1 = G4H * in_dim, n2 = G4H * H_DIM;
            cvt_f16_k<<<(n1 + 255) / 256, 256, 0, stream>>>(wih[l][d], WIH16[l][d], n1);
            cvt_f16_k<<<(n2 + 255) / 256, 256, 0, stream>>>(whh[l][d], WHH16[l][d], n2);
            bias_sum_k<<<2, 256, 0, stream>>>(bih[l][d], bhh[l][d], BIAS[l][d], G4H);
        }
    fcpad_k<<<(K_PAD * 2 * H_DIM + 255) / 256, 256, 0, stream>>>(fcw, fcb, FCW16, FCB16);

    // embedding gather (time-major f16)
    embed_k<<<(S_LEN * B_N * E_DIM) / 256, 256, 0, stream>>>(char_ids, emb, X16);

    // LDS: G f16 + h f16 + double-buffered x tile (always reserve TDM footprint)
    const size_t lds0 = (size_t)B_N * G4H * 2 + (size_t)B_N * H_DIM * 2 + 2u * B_N * E_DIM * 2;       // 224 KB
    const size_t lds1 = (size_t)B_N * G4H * 2 + (size_t)B_N * H_DIM * 2 + 2u * B_N * 2 * H_DIM * 2;   // 288 KB

    // layer 0: x [S,B,128] -> H0 [S,B,256]
    lstm_k<E_DIM><<<2, 1024, lds0, stream>>>(
        X16, WIH16[0][0], WHH16[0][0], BIAS[0][0],
             WIH16[0][1], WHH16[0][1], BIAS[0][1], H0);
    // layer 1: H0 [S,B,256] -> H1 [S,B,256]
    lstm_k<2 * H_DIM><<<2, 1024, lds1, stream>>>(
        H0, WIH16[1][0], WHH16[1][0], BIAS[1][0],
            WIH16[1][1], WHH16[1][1], BIAS[1][1], H1);

    // emissions: 131072 rows / 16 per wave = 8192 waves -> 1024 blocks x 8 waves
    emis_k<<<1024, 256, 0, stream>>>(H1, FCW16, FCB16, EM);

    // CRF NLL
    crf_k<<<1, 128, 0, stream>>>(EM, tags, mask, transm, startv, endv, (float*)d_out);
}
